// Mamba2Mixer_71038759076011
// MI455X (gfx1250) — matrix operand order repaired and checked by reading, not compiled, not run on hardware
//
#include <hip/hip_runtime.h>
#include <hip/hip_bf16.h>

// ---------------------------------------------------------------------------
// Mamba2 mixer forward, CDNA5 (gfx1250), wave32 + WMMA bf16 + TDM staging.
// B=2 L=4096 d_model=1024 d_inner=2048 H=16 ds=hd=128 chunk=128
// ---------------------------------------------------------------------------

typedef __bf16 bf16_t;
typedef __attribute__((ext_vector_type(16))) __bf16 v16bf;
typedef __attribute__((ext_vector_type(8)))  float  v8f;
typedef __attribute__((ext_vector_type(4)))  unsigned int v4u;
typedef __attribute__((ext_vector_type(8)))  int    v8i;
typedef __attribute__((ext_vector_type(4)))  int    v4i;

static __device__ __forceinline__ float  b2f(bf16_t x) { return (float)x; }
static __device__ __forceinline__ bf16_t f2b(float x)  { return (bf16_t)x; }

// D = A(16x32 bf16) * B(32x16 bf16) + C (f32), wave32
static __device__ __forceinline__ v8f wmma_bf16(v16bf a, v16bf b, v8f c) {
  return __builtin_amdgcn_wmma_f32_16x16x32_bf16(
      /*neg_a=*/false, a, /*neg_b=*/false, b,
      /*c_mod=*/(short)0, c, /*reuse_a=*/false, /*reuse_b=*/false);
}

// A-fragment: 16x32 (MxK). Lane L: row m0+(L&15); 16 contiguous K starting
// at k0+(L>>4)*16.
static __device__ __forceinline__ v16bf
load_frag_a(const bf16_t* __restrict__ base, int lda, int m0, int k0, int lane) {
  const bf16_t* p = base + (size_t)(m0 + (lane & 15)) * lda + k0 + ((lane >> 4) << 4);
  v16bf f;
  reinterpret_cast<float4*>(&f)[0] = *reinterpret_cast<const float4*>(p);
  reinterpret_cast<float4*>(&f)[1] = *reinterpret_cast<const float4*>(p + 8);
  return f;
}

// B-fragment: 32x16 (KxN) from a row-major [N x K] array ("W-like", NT gemm).
// Lane L: col n0+(L&15); 16 contiguous K starting at k0+(L>>4)*16.
static __device__ __forceinline__ v16bf
load_frag_b(const bf16_t* __restrict__ base, int ldw, int n0, int k0, int lane) {
  const bf16_t* p = base + (size_t)(n0 + (lane & 15)) * ldw + k0 + ((lane >> 4) << 4);
  v16bf f;
  reinterpret_cast<float4*>(&f)[0] = reinterpret_cast<const float4*>(p)[0];
  reinterpret_cast<float4*>(&f)[1] = reinterpret_cast<const float4*>(p)[1];
  return f;
}

// ---------------------------------------------------------------------------
// Tensor Data Mover: 2-D bf16 tile (tile_w x tile_h) from global -> LDS.
// D# packing per CDNA5 ISA ch.8 (group0: count/lds_addr/global_addr/type=2,
// group1: data_size=2B, dims/tile/stride). Generic->u32 pointer truncation
// yields the LDS byte offset (the low 32 bits of the flat address).
// ---------------------------------------------------------------------------
#if defined(__has_builtin)
#if __has_builtin(__builtin_amdgcn_tensor_load_to_lds)
#define USE_TDM 1
#endif
#endif
#ifndef USE_TDM
#define USE_TDM 0
#endif

static __device__ __forceinline__ void tdm_load_tile_2d(
    void* lds_ptr, const bf16_t* gptr,
    unsigned tile_w, unsigned tile_h, unsigned stride_elems) {
#if USE_TDM
  unsigned lds_byte = (unsigned)(uintptr_t)lds_ptr;
  unsigned long long ga = (unsigned long long)(uintptr_t)gptr;
  v4u g0;
  g0[0] = 1u;                                        // count=1 (valid, user mode)
  g0[1] = lds_byte;                                  // lds_addr
  g0[2] = (unsigned)ga;                              // global_addr[31:0]
  g0[3] = (unsigned)((ga >> 32) & 0x01FFFFFFu) | (2u << 30);  // addr[56:32] | type=2
  v8i g1;
  g1[0] = 0x10000;                                   // data_size=1 -> 2 bytes
  g1[1] = (int)((tile_w & 0xFFFFu) << 16);           // tensor_dim0[15:0]
  g1[2] = (int)((tile_w >> 16) | ((tile_h & 0xFFFFu) << 16));  // dim0 hi | dim1 lo
  g1[3] = (int)((tile_h >> 16) | (tile_w << 16));    // dim1 hi | tile_dim0
  g1[4] = (int)(tile_h & 0xFFFFu);                   // tile_dim1 (tile_dim2=0)
  g1[5] = (int)stride_elems;                         // tensor_dim0_stride[31:0]
  g1[6] = 0;                                         // stride0 hi | stride1 lo
  g1[7] = 0;
  v4i gz = {0, 0, 0, 0};
#if __clang_major__ >= 23
  v8i gz8 = {0, 0, 0, 0, 0, 0, 0, 0};
  __builtin_amdgcn_tensor_load_to_lds(g0, g1, gz, gz, gz8, 0);
#else
  __builtin_amdgcn_tensor_load_to_lds(g0, g1, gz, gz, 0);
#endif
#else
  (void)lds_ptr; (void)gptr; (void)tile_w; (void)tile_h; (void)stride_elems;
#endif
}

// ---------------------------------------------------------------------------
// fp32 -> bf16 conversion
// ---------------------------------------------------------------------------
__global__ void cvt_f32_to_bf16(const float* __restrict__ src,
                                bf16_t* __restrict__ dst, int n) {
  int i = blockIdx.x * blockDim.x + threadIdx.x;
  if (i < n) dst[i] = f2b(src[i]);
}

// ---------------------------------------------------------------------------
// NT GEMM epilogue modes
// ---------------------------------------------------------------------------
enum { EPI_SPLIT_VG = 0, EPI_BF16 = 1, EPI_F32 = 2 };

template <int EPI>
static __device__ __forceinline__ void gemm_epilogue(
    const v8f acc[2][4], int wm, int wn, int lane, int N,
    float* __restrict__ outF, bf16_t* __restrict__ outB, int dInner) {
#pragma unroll
  for (int i = 0; i < 2; ++i)
#pragma unroll
    for (int j = 0; j < 4; ++j)
#pragma unroll
      for (int v = 0; v < 8; ++v) {
        int row = wm + 16 * i + ((lane >> 4) << 3) + v;
        int col = wn + 16 * j + (lane & 15);
        float x = acc[i][j][v];
        if (EPI == EPI_F32) {
          outF[(size_t)row * N + col] = x;
        } else if (EPI == EPI_BF16) {
          outB[(size_t)row * N + col] = f2b(x);
        } else {  // split: cols < dInner -> value f32; rest -> silu(gate) bf16
          if (col < dInner) {
            outF[(size_t)row * dInner + col] = x;
          } else {
            float s = x / (1.0f + __expf(-x));
            outB[(size_t)row * dInner + (col - dInner)] = f2b(s);
          }
        }
      }
}

// ---------------------------------------------------------------------------
// TDM-staged NT GEMM: out[m,n] = sum_k A[m,k] * W[n,k]
// Block 256 thr = 8 waves, tile 128x128, BK=32, double-buffered LDS tiles
// moved by the Tensor Data Mover (wave 0 issues, TENSORcnt + barrier sync).
// ---------------------------------------------------------------------------
template <int EPI>
__global__ __launch_bounds__(256) void gemm_bf16_nt_tdm(
    const bf16_t* __restrict__ A, const bf16_t* __restrict__ W,
    int M, int N, int K,
    float* __restrict__ outF, bf16_t* __restrict__ outB, int dInner) {
  __shared__ bf16_t sA[2][128 * 32];
  __shared__ bf16_t sB[2][128 * 32];
  const int lane = threadIdx.x & 31;
  const int wave = threadIdx.x >> 5;
  const int bm = blockIdx.y * 128;
  const int bn = blockIdx.x * 128;
  const int wml = (wave & 3) * 32;   // wave row inside block tile
  const int wnl = (wave >> 2) * 64;  // wave col inside block tile

  v8f acc[2][4];
#pragma unroll
  for (int i = 0; i < 2; ++i)
#pragma unroll
    for (int j = 0; j < 4; ++j) acc[i][j] = (v8f){};

  const int KT = K / 32;
  const bf16_t* Abase = A + (size_t)bm * K;
  const bf16_t* Wbase = W + (size_t)bn * K;

  if (wave == 0) {
    tdm_load_tile_2d(&sA[0][0], Abase, 32, 128, (unsigned)K);
    tdm_load_tile_2d(&sB[0][0], Wbase, 32, 128, (unsigned)K);
  }

  for (int kt = 0; kt < KT; ++kt) {
    const int cur = kt & 1;
    if (wave == 0) {
      if (kt + 1 < KT) {
        const int nxt = cur ^ 1;
        tdm_load_tile_2d(&sA[nxt][0], Abase + (kt + 1) * 32, 32, 128, (unsigned)K);
        tdm_load_tile_2d(&sB[nxt][0], Wbase + (kt + 1) * 32, 32, 128, (unsigned)K);
        __builtin_amdgcn_s_wait_tensorcnt((short)2);  // current pair landed
      } else {
        __builtin_amdgcn_s_wait_tensorcnt((short)0);
      }
    }
    __syncthreads();  // publish LDS tiles to all 8 waves

    v16bf a0 = load_frag_a(&sA[cur][0], 32, wml, 0, lane);
    v16bf a1 = load_frag_a(&sA[cur][0], 32, wml + 16, 0, lane);
    v16bf b0 = load_frag_b(&sB[cur][0], 32, wnl, 0, lane);
    v16bf b1 = load_frag_b(&sB[cur][0], 32, wnl + 16, 0, lane);
    v16bf b2 = load_frag_b(&sB[cur][0], 32, wnl + 32, 0, lane);
    v16bf b3 = load_frag_b(&sB[cur][0], 32, wnl + 48, 0, lane);
    acc[0][0] = wmma_bf16(a0, b0, acc[0][0]);
    acc[0][1] = wmma_bf16(a0, b1, acc[0][1]);
    acc[0][2] = wmma_bf16(a0, b2, acc[0][2]);
    acc[0][3] = wmma_bf16(a0, b3, acc[0][3]);
    acc[1][0] = wmma_bf16(a1, b0, acc[1][0]);
    acc[1][1] = wmma_bf16(a1, b1, acc[1][1]);
    acc[1][2] = wmma_bf16(a1, b2, acc[1][2]);
    acc[1][3] = wmma_bf16(a1, b3, acc[1][3]);

    __syncthreads();  // tiles consumed; safe to overwrite next iteration
  }

  gemm_epilogue<EPI>(acc, bm + wml, bn + wnl, lane, N, outF, outB, dInner);
}

// ---------------------------------------------------------------------------
// Fallback NT GEMM (direct global fragment loads) — used only if the TDM
// builtin is unavailable on this toolchain.
// ---------------------------------------------------------------------------
template <int EPI>
__global__ __launch_bounds__(256) void gemm_bf16_nt(
    const bf16_t* __restrict__ A, const bf16_t* __restrict__ W,
    int M, int N, int K,
    float* __restrict__ outF, bf16_t* __restrict__ outB, int dInner) {
  const int lane = threadIdx.x & 31;
  const int wave = threadIdx.x >> 5;
  const int wm = blockIdx.y * 128 + (wave & 3) * 32;
  const int wn = blockIdx.x * 128 + (wave >> 2) * 64;

  v8f acc[2][4];
#pragma unroll
  for (int i = 0; i < 2; ++i)
#pragma unroll
    for (int j = 0; j < 4; ++j) acc[i][j] = (v8f){};

  for (int k0 = 0; k0 < K; k0 += 32) {
    v16bf a0 = load_frag_a(A, K, wm, k0, lane);
    v16bf a1 = load_frag_a(A, K, wm + 16, k0, lane);
    v16bf b0 = load_frag_b(W, K, wn, k0, lane);
    v16bf b1 = load_frag_b(W, K, wn + 16, k0, lane);
    v16bf b2 = load_frag_b(W, K, wn + 32, k0, lane);
    v16bf b3 = load_frag_b(W, K, wn + 48, k0, lane);
    acc[0][0] = wmma_bf16(a0, b0, acc[0][0]);
    acc[0][1] = wmma_bf16(a0, b1, acc[0][1]);
    acc[0][2] = wmma_bf16(a0, b2, acc[0][2]);
    acc[0][3] = wmma_bf16(a0, b3, acc[0][3]);
    acc[1][0] = wmma_bf16(a1, b0, acc[1][0]);
    acc[1][1] = wmma_bf16(a1, b1, acc[1][1]);
    acc[1][2] = wmma_bf16(a1, b2, acc[1][2]);
    acc[1][3] = wmma_bf16(a1, b3, acc[1][3]);
  }
  gemm_epilogue<EPI>(acc, wm, wn, lane, N, outF, outB, dInner);
}

// ---------------------------------------------------------------------------
// Causal depthwise conv (K=4) + bias + silu, fp32 in -> bf16 out
// ---------------------------------------------------------------------------
__global__ __launch_bounds__(256) void conv_silu_kernel(
    const float* __restrict__ valpre, const float* __restrict__ conv_w,
    const float* __restrict__ conv_b, bf16_t* __restrict__ value,
    int Bb, int L, int C, int Kc) {
  int idx = blockIdx.x * blockDim.x + threadIdx.x;
  if (idx >= Bb * L * C) return;
  int c = idx % C;
  int l = (idx / C) % L;
  int b = idx / (C * L);
  float acc = conv_b[c];
#pragma unroll
  for (int k = 0; k < 4; ++k) {
    int ls = l - (Kc - 1) + k;
    if (ls >= 0) acc += valpre[((size_t)b * L + ls) * C + c] * conv_w[c * Kc + k];
  }
  float s = acc / (1.0f + __expf(-acc));
  value[idx] = f2b(s);
}

// ---------------------------------------------------------------------------
// Scan stage 1: statesT[d][s] = sum_j x_dt[j,d]*B_w[j,s] via WMMA, with
// x_dt^T and (B*decay_rev)^T staged in LDS.
// ---------------------------------------------------------------------------
__global__ __launch_bounds__(256) void scan_states_kernel(
    const bf16_t* __restrict__ value, const bf16_t* __restrict__ Bm,
    const float* __restrict__ dt_bias, const float* __restrict__ A_log,
    float* __restrict__ statesT, int L, int H) {
  __shared__ bf16_t sXdtT[128 * 128];  // [d][j]
  __shared__ bf16_t sBT[128 * 128];    // [s][j], decay_rev applied
  const int n = blockIdx.x, h = blockIdx.y, b = blockIdx.z;
  const int nc = gridDim.x;
  const int tid = threadIdx.x;
  const float dt = log1pf(__expf(dt_bias[h]));
  const float lnDec = -__expf(A_log[h]) * dt;
  const size_t tok0 = (size_t)b * L + (size_t)n * 128;
  const int DIN = H * 128;

  for (int it = 0; it < 64; ++it) {
    int e = tid + 256 * it;
    int j = e >> 7, c = e & 127;
    size_t g = (tok0 + j) * DIN + h * 128 + c;
    sXdtT[c * 128 + j] = f2b(b2f(value[g]) * dt);
    sBT[c * 128 + j] = f2b(b2f(Bm[g]) * __expf(lnDec * (float)(127 - j)));
  }
  __syncthreads();

  const int lane = tid & 31, wave = tid >> 5;
  const int d0 = wave * 16;
  float* outBase = statesT + (size_t)((b * nc + n) * H + h) * (128 * 128);
  for (int st = 0; st < 8; ++st) {
    v8f acc = (v8f){};
#pragma unroll
    for (int k0 = 0; k0 < 128; k0 += 32) {
      v16bf a = load_frag_a(sXdtT, 128, d0, k0, lane);
      v16bf bb = load_frag_b(sBT, 128, st * 16, k0, lane);
      acc = wmma_bf16(a, bb, acc);
    }
#pragma unroll
    for (int v = 0; v < 8; ++v) {
      int dd = d0 + ((lane >> 4) << 3) + v;
      int ss = st * 16 + (lane & 15);
      outBase[dd * 128 + ss] = acc[v];
    }
  }
}

// ---------------------------------------------------------------------------
// Inter-chunk carry recurrence (per b,h): fp32 statesT -> bf16 carryT, 64 fp32 regs per thread.
// ---------------------------------------------------------------------------
__global__ __launch_bounds__(256) void carry_kernel(
    const float* __restrict__ statesT, bf16_t* __restrict__ carryT,
    const float* __restrict__ dt_bias, const float* __restrict__ A_log,
    int nc, int H) {
  const int h = blockIdx.x, b = blockIdx.y;
  const int tid = threadIdx.x;
  const float dt = log1pf(__expf(dt_bias[h]));
  const float lnDec = -__expf(A_log[h]) * dt;
  const float cd = __expf(lnDec * 128.0f);
  float r[64];
#pragma unroll
  for (int t = 0; t < 64; ++t) r[t] = 0.0f;
  for (int n = 0; n < nc; ++n) {
    size_t base = (size_t)((b * nc + n) * H + h) * (128 * 128);
#pragma unroll
    for (int t = 0; t < 64; ++t) {
      int e = tid + 256 * t;
      float s = statesT[base + e];
      carryT[base + e] = f2b(r[t]);
      r[t] = cd * r[t] + s;
    }
  }
}

// ---------------------------------------------------------------------------
// Scan stage 2: masked CB -> y_diag, y_off from carry; epilogue adds D*value,
// multiplies silu(gate), writes bf16 y for the output projection.
// ---------------------------------------------------------------------------
__global__ __launch_bounds__(256) void scan_y_kernel(
    const bf16_t* __restrict__ value, const bf16_t* __restrict__ Bm,
    const bf16_t* __restrict__ Cm, const bf16_t* __restrict__ carryT,
    const bf16_t* __restrict__ gateS, const float* __restrict__ dt_bias,
    const float* __restrict__ A_log, const float* __restrict__ D_param,
    bf16_t* __restrict__ yout, int L, int H) {
  __shared__ bf16_t sXdtT[128 * 128];  // [d][j]
  __shared__ bf16_t sCBL[128 * 128];   // [i][j] masked CB
  const int n = blockIdx.x, h = blockIdx.y, b = blockIdx.z;
  const int nc = gridDim.x;
  const int tid = threadIdx.x, lane = tid & 31, wave = tid >> 5;
  const float dt = log1pf(__expf(dt_bias[h]));
  const float lnDec = -__expf(A_log[h]) * dt;
  const float Dp = D_param[h];
  const size_t tok0 = (size_t)b * L + (size_t)n * 128;
  const int DIN = H * 128;
  const bf16_t* Cbase = Cm + tok0 * DIN + h * 128;
  const bf16_t* Bbase = Bm + tok0 * DIN + h * 128;
  const bf16_t* carryBase = carryT + (size_t)((b * nc + n) * H + h) * (128 * 128);

  for (int it = 0; it < 64; ++it) {
    int e = tid + 256 * it;
    int j = e >> 7, c = e & 127;
    sXdtT[c * 128 + j] = f2b(b2f(value[(tok0 + j) * DIN + h * 128 + c]) * dt);
  }
  __syncthreads();

  const int i0 = wave * 16;
  for (int jt = 0; jt < 8; ++jt) {
    v8f acc = (v8f){};
#pragma unroll
    for (int k0 = 0; k0 < 128; k0 += 32) {
      v16bf a = load_frag_a(Cbase, DIN, i0, k0, lane);
      v16bf bb = load_frag_b(Bbase, DIN, jt * 16, k0, lane);
      acc = wmma_bf16(a, bb, acc);
    }
#pragma unroll
    for (int v = 0; v < 8; ++v) {
      int i = i0 + ((lane >> 4) << 3) + v;
      int j = jt * 16 + (lane & 15);
      int diff = i - j;
      float val = (diff >= 0) ? acc[v] * __expf(lnDec * (float)diff) : 0.0f;
      sCBL[i * 128 + j] = f2b(val);
    }
  }
  __syncthreads();

  for (int dtl = 0; dtl < 8; ++dtl) {
    int dn0 = dtl * 16;
    v8f accD = (v8f){};
    v8f accO = (v8f){};
#pragma unroll
    for (int k0 = 0; k0 < 128; k0 += 32) {  // y_diag = (CB.L) @ x_dt
      v16bf a = load_frag_a(sCBL, 128, i0, k0, lane);
      v16bf bb = load_frag_b(sXdtT, 128, dn0, k0, lane);
      accD = wmma_bf16(a, bb, accD);
    }
#pragma unroll
    for (int s0 = 0; s0 < 128; s0 += 32) {  // y_off = C @ carry
      v16bf a = load_frag_a(Cbase, DIN, i0, s0, lane);
      v16bf bb = load_frag_b(carryBase, 128, dn0, s0, lane);
      accO = wmma_bf16(a, bb, accO);
    }
#pragma unroll
    for (int v = 0; v < 8; ++v) {
      int i = i0 + ((lane >> 4) << 3) + v;
      int d = dn0 + (lane & 15);
      size_t g = (tok0 + i) * DIN + h * 128 + d;
      float y = accD[v] + accO[v] * __expf(lnDec * (float)(i + 1));
      y += Dp * b2f(value[g]);
      y *= b2f(gateS[g]);
      yout[g] = f2b(y);
    }
  }
}

// ---------------------------------------------------------------------------
// Host launcher
// ---------------------------------------------------------------------------
extern "C" void kernel_launch(void* const* d_in, const int* in_sizes, int n_in,
                              void* d_out, int out_size, void* d_ws, size_t ws_size,
                              hipStream_t stream) {
  (void)in_sizes; (void)n_in; (void)out_size; (void)ws_size;
  const int Bb = 2, L = 4096, d_model = 1024, d_inner = 2048, H = 16, Kc = 4;
  const int M = Bb * L;      // 8192 tokens
  const int nc = L / 128;    // 32 chunks

  const float* x          = (const float*)d_in[0];
  const float* x_prenorm  = (const float*)d_in[1];
  const float* in_proj_w  = (const float*)d_in[2];
  const float* conv_w     = (const float*)d_in[3];
  const float* conv_b     = (const float*)d_in[4];
  const float* B_proj_w   = (const float*)d_in[5];
  const float* C_proj_w   = (const float*)d_in[6];
  const float* dt_bias    = (const float*)d_in[7];
  const float* A_log      = (const float*)d_in[8];
  const float* D_param    = (const float*)d_in[9];
  const float* out_proj_w = (const float*)d_in[10];

  char* base = (char*)d_ws;
  size_t off = 0;
  auto carve = [&](size_t bytes) -> void* {
    void* p = base + off;
    off += (bytes + 255) & ~(size_t)255;
    return p;
  };

  bf16_t* xb      = (bf16_t*)carve((size_t)M * d_model * 2);
  bf16_t* xpb     = (bf16_t*)carve((size_t)M * d_model * 2);
  bf16_t* w_in_b  = (bf16_t*)carve((size_t)2 * d_inner * d_model * 2);
  bf16_t* wBb     = (bf16_t*)carve((size_t)d_inner * d_model * 2);
  bf16_t* wCb     = (bf16_t*)carve((size_t)d_inner * d_model * 2);
  bf16_t* w_out_b = (bf16_t*)carve((size_t)d_model * d_inner * 2);
  float*  valpre  = (float*)carve((size_t)M * d_inner * 4);  // reused as statesT
  bf16_t* gateS   = (bf16_t*)carve((size_t)M * d_inner * 2);
  bf16_t* valueb  = (bf16_t*)carve((size_t)M * d_inner * 2);
  bf16_t* Bmb     = (bf16_t*)carve((size_t)M * d_inner * 2);
  bf16_t* Cmb     = (bf16_t*)carve((size_t)M * d_inner * 2);
  bf16_t* carryTb = (bf16_t*)carve((size_t)Bb * nc * H * 128 * 128 * 2);
  float*  statesT = valpre;  // alias: valpre dead after conv (same 64 MB)
  bf16_t* yb      = xb;      // alias: xb+xpb dead after projections

  auto cvt = [&](const float* s, bf16_t* d, int n) {
    cvt_f32_to_bf16<<<(n + 255) / 256, 256, 0, stream>>>(s, d, n);
  };
  cvt(x,          xb,      M * d_model);
  cvt(x_prenorm,  xpb,     M * d_model);
  cvt(in_proj_w,  w_in_b,  2 * d_inner * d_model);
  cvt(B_proj_w,   wBb,     d_inner * d_model);
  cvt(C_proj_w,   wCb,     d_inner * d_model);
  cvt(out_proj_w, w_out_b, d_model * d_inner);

#if USE_TDM
#define GEMM_SPLIT gemm_bf16_nt_tdm<EPI_SPLIT_VG>
#define GEMM_B16   gemm_bf16_nt_tdm<EPI_BF16>
#define GEMM_F32   gemm_bf16_nt_tdm<EPI_F32>
#else
#define GEMM_SPLIT gemm_bf16_nt<EPI_SPLIT_VG>
#define GEMM_B16   gemm_bf16_nt<EPI_BF16>
#define GEMM_F32   gemm_bf16_nt<EPI_F32>
#endif

  // in_proj: value (f32) + silu(gate) (bf16)
  dim3 g1(2 * d_inner / 128, M / 128);
  GEMM_SPLIT<<<g1, 256, 0, stream>>>(xb, w_in_b, M, 2 * d_inner, d_model,
                                     valpre, gateS, d_inner);

  // B/C projections (bf16 out)
  dim3 g2(d_inner / 128, M / 128);
  GEMM_B16<<<g2, 256, 0, stream>>>(xpb, wBb, M, d_inner, d_model, nullptr, Bmb, 0);
  GEMM_B16<<<g2, 256, 0, stream>>>(xpb, wCb, M, d_inner, d_model, nullptr, Cmb, 0);

  // depthwise conv + silu
  conv_silu_kernel<<<((size_t)M * d_inner + 255) / 256, 256, 0, stream>>>(
      valpre, conv_w, conv_b, valueb, Bb, L, d_inner, Kc);

  // chunked selective scan
  dim3 gs(nc, H, Bb);
  scan_states_kernel<<<gs, 256, 0, stream>>>(valueb, Bmb, dt_bias, A_log,
                                             statesT, L, H);
  carry_kernel<<<dim3(H, Bb), 256, 0, stream>>>(statesT, carryTb, dt_bias,
                                                A_log, nc, H);
  scan_y_kernel<<<gs, 256, 0, stream>>>(valueb, Bmb, Cmb, carryTb, gateS,
                                        dt_bias, A_log, D_param, yb, L, H);

  // out_proj (f32 out)
  dim3 g4(d_model / 128, M / 128);
  GEMM_F32<<<g4, 256, 0, stream>>>(yb, w_out_b, M, d_model, d_inner,
                                   (float*)d_out, nullptr, 0);
}
